// GlobalLSTMCell_60206851555531
// MI455X (gfx1250) — matrix operand order repaired and checked by reading, not compiled, not run on hardware
//
#include <hip/hip_runtime.h>
#include <stdint.h>

// ---------------- problem dims ----------------
#define BATCH 4096
#define FEAT  1024
#define HID   1024
#define KDIM  (FEAT + HID)        // 2048
#define NGATE 4

// ---------------- GEMM tiling ----------------
#define BLOCK_M  128
#define BLOCK_N  32               // hidden columns per block (x 4 gates)
#define KSTEP    32
#define KSTEPS   (KDIM / KSTEP)   // 64
#define NTHREADS 256
#define APAD     40               // padded LDS row length (bf16) -> 80B stride, bank-conflict free

typedef __bf16 bf16;
typedef bf16  v8bf  __attribute__((ext_vector_type(8)));
typedef bf16  v16bf __attribute__((ext_vector_type(16)));
typedef float v8f   __attribute__((ext_vector_type(8)));
typedef int   v4i   __attribute__((ext_vector_type(4)));

// ---- CDNA5 async global->LDS copy path (guarded; falls back to ld+ds_store) ----
#if defined(__has_builtin)
# if __has_builtin(__builtin_amdgcn_global_load_async_to_lds_b128)
#  define HAVE_ASYNC_LDS 1
# endif
#endif
#ifndef HAVE_ASYNC_LDS
# define HAVE_ASYNC_LDS 0
#endif

#if HAVE_ASYNC_LDS
# if __has_builtin(__builtin_amdgcn_s_wait_asynccnt)
#  define WAIT_ASYNC(n) __builtin_amdgcn_s_wait_asynccnt(n)
# else
#  define WAIT_ASYNC(n) asm volatile("s_wait_asynccnt %0" :: "n"(n) : "memory")
# endif
#else
# define WAIT_ASYNC(n) ((void)0)
#endif

__device__ __forceinline__ void copy16(bf16* dst_lds, const bf16* src) {
#if HAVE_ASYNC_LDS
  // Generic LDS pointers carry the LDS byte offset in the low 32 bits (aperture
  // rule), so an inttoptr to AS(3) is the LDS address.
  __builtin_amdgcn_global_load_async_to_lds_b128(
      (__attribute__((address_space(1))) v4i*)(uintptr_t)src,
      (__attribute__((address_space(3))) v4i*)(uint32_t)(uintptr_t)dst_lds,
      0, 0);
#else
  *(v8bf*)dst_lds = *(const v8bf*)src;
#endif
}

// Fast, saturation-safe activations built on v_exp_f32:
//   sigmoid(x) = 1/(1+e^-x);  tanh(x) = 2*sigmoid(2x)-1 = 2/(1+e^-2x)-1
// Both have correct asymptotes (exp underflows to 0 / overflows to +inf cleanly).
__device__ __forceinline__ float fast_sigmoid(float x) {
  return 1.0f / (1.0f + __expf(-x));
}
__device__ __forceinline__ float fast_tanh(float x) {
  return 2.0f / (1.0f + __expf(-2.0f * x)) - 1.0f;
}

// ---------------- pack kernels (fp32 -> bf16, one-time, L2-resident afterwards) ----------------
__global__ void pack_x_kernel(const float* __restrict__ x,
                              const float* __restrict__ h_prev,
                              bf16* __restrict__ Xbf) {
  int idx = blockIdx.x * blockDim.x + threadIdx.x;   // 0 .. BATCH*FEAT-1
  int b = idx >> 10;
  int k = idx & 1023;
  Xbf[(size_t)b * KDIM + k]        = (bf16)x[idx];
  Xbf[(size_t)b * KDIM + FEAT + k] = (bf16)h_prev[idx];
}

// Wbf layout: [gate][n (HID)][k (KDIM)] with k contiguous (K-major per output column).
__global__ void pack_w_kernel(const float* __restrict__ W_ifo_x,
                              const float* __restrict__ W_ifo_h,
                              const float* __restrict__ W_b_x,
                              const float* __restrict__ W_b_h,
                              bf16* __restrict__ Wbf) {
  int idx = blockIdx.x * blockDim.x + threadIdx.x;   // 0 .. 4*HID*KDIM-1
  int k  = idx & (KDIM - 1);
  int n  = (idx >> 11) & (HID - 1);
  int g  = idx >> 21;
  int kk = k & (FEAT - 1);
  float v;
  if (g < 3) {
    const float* W = (k < FEAT) ? W_ifo_x : W_ifo_h;
    v = W[(size_t)kk * (3 * HID) + g * HID + n];
  } else {
    const float* W = (k < FEAT) ? W_b_x : W_b_h;
    v = W[(size_t)kk * HID + n];
  }
  Wbf[idx] = (bf16)v;
}

// ---------------- fused WMMA GEMM + LSTM gate epilogue ----------------
__global__ __launch_bounds__(NTHREADS, 2)
void lstm_wmma_gemm(const bf16* __restrict__ Xbf,    // [BATCH][KDIM]
                    const bf16* __restrict__ Wbf,    // [4][HID][KDIM]
                    const float* __restrict__ c_prev,
                    const float* __restrict__ b_ifo_x,
                    const float* __restrict__ b_ifo_h,
                    const float* __restrict__ b_b_x,
                    const float* __restrict__ b_b_h,
                    float* __restrict__ out_h,
                    float* __restrict__ out_c) {
  __shared__ bf16 Alds[2][BLOCK_M][APAD];
  __shared__ bf16 Blds[2][NGATE][BLOCK_N][APAD];

  const int tid  = threadIdx.x;
  const int bm   = blockIdx.x;
  const int bn   = blockIdx.y;
  const int wave = tid >> 5;
  const int lane = tid & 31;
  const int wm   = wave >> 1;      // 0..3 : 32-row slice of the 128-row tile
  const int wn   = wave & 1;       // 0..1 : 16-col slice of the 32-col tile
  const int half = lane >> 4;      // 0/1
  const int r    = lane & 15;

  const bf16* Ag = Xbf + (size_t)bm * BLOCK_M * KDIM;

  v8f acc[2][NGATE] = {};

  auto fill = [&](int buf, int k0) {
    // A tile: 128 rows x 32 bf16 = 512 chunks of 16B
#pragma unroll
    for (int j = 0; j < 2; ++j) {
      int c   = tid + j * NTHREADS;
      int row = c >> 2;
      int off = (c & 3) * 8;
      copy16(&Alds[buf][row][off], Ag + (size_t)row * KDIM + k0 + off);
    }
    // B tiles: 4 gates x 32 cols x 32 bf16 = 512 chunks of 16B
#pragma unroll
    for (int j = 0; j < 2; ++j) {
      int c   = tid + j * NTHREADS;
      int g   = c >> 7;
      int rem = c & 127;
      int n   = rem >> 2;
      int off = (rem & 3) * 8;
      const bf16* src = Wbf + ((size_t)g * HID + (size_t)bn * BLOCK_N + n) * KDIM + k0 + off;
      copy16(&Blds[buf][g][n][off], src);
    }
  };

  fill(0, 0);

  for (int s = 0; s < KSTEPS; ++s) {
    const int cur = s & 1;
    if (s + 1 < KSTEPS) {
      fill(cur ^ 1, (s + 1) * KSTEP);
      WAIT_ASYNC(4);   // each thread issued 4 chunks for the NEXT buffer; <=4 => current done
    } else {
      WAIT_ASYNC(0);
    }
    __syncthreads();

    // A fragments: 16x32 bf16, ISA lane layout:
    //   lanes 0-15  : row r, K 0-7 (v0-3) and K 16-23 (v4-7)
    //   lanes 16-31 : row r, K 8-15 and K 24-31
    v16bf af[2];
#pragma unroll
    for (int m = 0; m < 2; ++m) {
      const bf16* ap = &Alds[cur][wm * 32 + m * 16 + r][half * 8];
      v8bf lo = *(const v8bf*)ap;
      v8bf hi = *(const v8bf*)(ap + 16);
      v16bf v;
#pragma unroll
      for (int i = 0; i < 8; ++i) { v[i] = lo[i]; v[i + 8] = hi[i]; }
      af[m] = v;
    }

    // B fragments: 32x16 bf16, lane n holds column n; lanes 0-15 K 0-7 and K 16-23, lanes 16-31 K 8-15 and K 24-31
    v16bf bfr[NGATE];
#pragma unroll
    for (int g = 0; g < NGATE; ++g) {
      const bf16* bp = &Blds[cur][g][wn * 16 + r][half * 8];
      v8bf lo = *(const v8bf*)bp;
      v8bf hi = *(const v8bf*)(bp + 16);
      v16bf v;
#pragma unroll
      for (int i = 0; i < 8; ++i) { v[i] = lo[i]; v[i + 8] = hi[i]; }
      bfr[g] = v;
    }

#pragma unroll
    for (int m = 0; m < 2; ++m)
#pragma unroll
      for (int g = 0; g < NGATE; ++g)
        acc[m][g] = __builtin_amdgcn_wmma_f32_16x16x32_bf16(
            false, af[m], false, bfr[g], (short)0, acc[m][g], false, false);

    __syncthreads();
  }

  // ---- fused LSTM epilogue (C/D layout: lane -> N = lane%16, VGPR e -> M = e + 8*(lane/16)) ----
  const int col = bn * BLOCK_N + wn * 16 + r;
  const float bi  = b_ifo_x[col]           + b_ifo_h[col];
  const float bff = b_ifo_x[col + HID]     + b_ifo_h[col + HID];
  const float bo  = b_ifo_x[col + 2 * HID] + b_ifo_h[col + 2 * HID];
  const float ba  = b_b_x[col]             + b_b_h[col];

#pragma unroll
  for (int m = 0; m < 2; ++m) {
#pragma unroll
    for (int e = 0; e < 8; ++e) {
      const int    row = bm * BLOCK_M + wm * 32 + m * 16 + half * 8 + e;
      const size_t o   = (size_t)row * HID + col;
      const float iv = fast_sigmoid(acc[m][0][e] + bi);
      const float fv = fast_sigmoid(acc[m][1][e] + bff);
      const float ov = fast_sigmoid(acc[m][2][e] + bo);
      const float av = fast_tanh(acc[m][3][e] + ba);
      const float cc = iv * av + fv * c_prev[o];
      out_c[o] = cc;
      out_h[o] = ov * fast_tanh(cc);
    }
  }
}

extern "C" void kernel_launch(void* const* d_in, const int* in_sizes, int n_in,
                              void* d_out, int out_size, void* d_ws, size_t ws_size,
                              hipStream_t stream) {
  (void)in_sizes; (void)n_in; (void)out_size; (void)ws_size;

  const float* x       = (const float*)d_in[0];
  const float* h_prev  = (const float*)d_in[1];
  const float* c_prev  = (const float*)d_in[2];
  const float* W_ifo_x = (const float*)d_in[3];
  const float* b_ifo_x = (const float*)d_in[4];
  const float* W_ifo_h = (const float*)d_in[5];
  const float* b_ifo_h = (const float*)d_in[6];
  const float* W_b_x   = (const float*)d_in[7];
  const float* b_b_x   = (const float*)d_in[8];
  const float* W_b_h   = (const float*)d_in[9];
  const float* b_b_h   = (const float*)d_in[10];

  bf16* Xbf = (bf16*)d_ws;                                                  // 16 MB
  bf16* Wbf = (bf16*)((char*)d_ws + (size_t)BATCH * KDIM * sizeof(bf16));   // +16 MB

  float* out_h = (float*)d_out;
  float* out_c = out_h + (size_t)BATCH * HID;

  pack_x_kernel<<<(BATCH * FEAT) / NTHREADS, NTHREADS, 0, stream>>>(x, h_prev, Xbf);
  pack_w_kernel<<<(NGATE * HID * KDIM) / NTHREADS, NTHREADS, 0, stream>>>(
      W_ifo_x, W_ifo_h, W_b_x, W_b_h, Wbf);

  dim3 grid(BATCH / BLOCK_M, HID / BLOCK_N);
  lstm_wmma_gemm<<<grid, NTHREADS, 0, stream>>>(
      Xbf, Wbf, c_prev, b_ifo_x, b_ifo_h, b_b_x, b_b_h, out_h, out_c);
}
